// SentenceEmbeddingSelfAttention_22849226015195
// MI455X (gfx1250) — matrix operand order repaired and checked by reading, not compiled, not run on hardware
//
#include <hip/hip_runtime.h>

// ---------------- problem constants ----------------
static constexpr int B = 128, T = 512, E = 256, H = 256, DA = 25, HEADS = 5;
static constexpr int G4 = 4 * H;          // 1024 gate width
static constexpr float NEGINF = -1e30f;

#define DEV __device__ __forceinline__

// ---------------- WMMA types / helpers ----------------
typedef __attribute__((ext_vector_type(16))) __bf16   v16bf;
typedef __attribute__((ext_vector_type(8)))  float    v8f;
typedef __attribute__((ext_vector_type(4)))  unsigned int u32x4;

union FragU { v16bf v; u32x4 q[2]; unsigned short u[16]; };

DEV unsigned short f2bf(float f) {
    unsigned int u = __float_as_uint(f);
    u += 0x7FFFu + ((u >> 16) & 1u);         // round-to-nearest-even
    return (unsigned short)(u >> 16);
}

// A fragment (16x32 bf16, row-major source with row stride ld):
// lane L: M = L&15, K in {kb..kb+7, kb+16..kb+23} with kb = (L>=16)*8
DEV v16bf loadA_rm(const unsigned short* base, int ld) {
    const int lane = threadIdx.x & 31;
    const int m  = lane & 15;
    const int kb = (lane >> 4) * 8;
    const unsigned short* p = base + (size_t)m * ld + kb;
    FragU r;
    r.q[0] = *(const u32x4*)(p);
    r.q[1] = *(const u32x4*)(p + 16);
    return r.v;
}

// B fragment (32x16 bf16) where B[k][n] = W[n0+n][k0+k], W row-major stride ld.
// lane L: n = L&15, K in {kb..kb+7, kb+16..kb+23} with kb = (L>=16)*8, from one W row.
DEV v16bf loadB_wt(const unsigned short* W, int ld, int n0, int k0) {
    const int lane = threadIdx.x & 31;
    const int n  = lane & 15;
    const int kb = (lane >> 4) * 8;
    const unsigned short* p = W + (size_t)(n0 + n) * ld + k0 + kb;
    FragU r;
    r.q[0] = *(const u32x4*)(p);
    r.q[1] = *(const u32x4*)(p + 16);
    return r.v;
}

DEV v8f wmma_bf16(v16bf a, v16bf b, v8f c) {
    return __builtin_amdgcn_wmma_f32_16x16x32_bf16(
        /*neg_a=*/false, a, /*neg_b=*/false, b,
        /*c_mod=*/(short)0, c, /*reuse_a=*/false, /*reuse_b=*/false);
}

DEV float sigm(float x) { return 1.0f / (1.0f + __expf(-x)); }

// ---------------- kernels ----------------

// Gather embeddings for x and per-sample-reversed x_rev, cast to bf16.
__global__ void prep_embed(const int* __restrict__ wid, const int* __restrict__ len,
                           const float* __restrict__ emb,
                           unsigned short* __restrict__ xf, unsigned short* __restrict__ xr) {
    int bt = blockIdx.x;                 // B*T blocks
    int b = bt / T, t = bt % T;
    int e = threadIdx.x;                 // E = 256 threads
    int L  = len[b];
    int rt = L - 1 - t; if (rt < 0) rt = 0;
    int wf = wid[b * T + t];
    int wr = wid[b * T + rt];
    xf[(size_t)bt * E + e] = f2bf(emb[(size_t)wf * E + e]);
    xr[(size_t)bt * E + e] = f2bf(emb[(size_t)wr * E + e]);
}

__global__ void conv_bf16(const float* __restrict__ src, unsigned short* __restrict__ dst, int n) {
    int i = blockIdx.x * blockDim.x + threadIdx.x;
    if (i < n) dst[i] = f2bf(src[i]);
}

// px[t][b][g] = sum_k X[b*T+t][k] * W[g][k] + bias[g]
// One wave per (16 x 128) output slab: 8 N-tiles share each A fragment,
// so every 32-byte A load feeds 8 chained bf16 WMMAs (8x less x traffic).
__global__ __launch_bounds__(256)
void gemm_px(const unsigned short* __restrict__ X, const unsigned short* __restrict__ W,
             const float* __restrict__ bias, float* __restrict__ P) {
    const int wave = blockIdx.x * (blockDim.x >> 5) + (threadIdx.x >> 5);
    const int mt   = wave >> 3;          // 8 waves per M tile (NT=64 -> 8 tiles each)
    const int ntb  = (wave & 7) * 8;     // first N tile of this wave
    const int lane = threadIdx.x & 31;
    const int nlo  = lane & 15;

    v8f acc[8];
#pragma unroll
    for (int q = 0; q < 8; ++q) {
        float bv = bias[(ntb + q) * 16 + nlo];
        acc[q] = (v8f){ bv, bv, bv, bv, bv, bv, bv, bv };
    }

    const unsigned short* Xtile = X + (size_t)(mt * 16) * E;
    for (int kk = 0; kk < E; kk += 32) {
        v16bf a = loadA_rm(Xtile + kk, E);
#pragma unroll
        for (int q = 0; q < 8; ++q) {
            v16bf b = loadB_wt(W, E, (ntb + q) * 16, kk);
            acc[q] = wmma_bf16(a, b, acc[q]);
        }
    }

    const int mhi = (lane >> 4) * 8;
#pragma unroll
    for (int q = 0; q < 8; ++q) {
        const int nc = (ntb + q) * 16 + nlo;
#pragma unroll
        for (int i = 0; i < 8; ++i) {
            int row  = mt * 16 + mhi + i;       // row = b*T + t
            int bidx = row / T;
            int t    = row % T;
            P[((size_t)t * B + bidx) * G4 + nc] = acc[q][i];
        }
    }
}

// Sequential LSTM scan. grid = (B/16, 2 directions), 256 threads = 8 waves.
// Wave w owns hidden units [32w, 32w+32): 4 gates x 2 j-tiles = 8 WMMA tiles.
// h lives in LDS as bf16 between timesteps (workgroup barrier per step).
__global__ __launch_bounds__(256)
void lstm_scan(const float* __restrict__ pxF, const float* __restrict__ pxB,
               const unsigned short* __restrict__ WhhF, const unsigned short* __restrict__ WhhB,
               float* __restrict__ hF, float* __restrict__ hB) {
    const int dir = blockIdx.y;
    const float* px          = dir ? pxB  : pxF;
    const unsigned short* Wh = dir ? WhhB : WhhF;
    float* hout              = dir ? hB   : hF;

    const int bbase = blockIdx.x * 16;
    const int wv    = threadIdx.x >> 5;
    const int lane  = threadIdx.x & 31;
    const int jbase = wv * 32;
    const int nlo   = lane & 15;
    const int mhi   = (lane >> 4) * 8;

    __shared__ unsigned short hsh[16][H + 8];   // +8 u16 pad = 16B, keeps alignment

    for (int idx = threadIdx.x; idx < 16 * (H + 8); idx += blockDim.x)
        (&hsh[0][0])[idx] = 0;
    __syncthreads();

    float creg[2][8];
#pragma unroll
    for (int jt = 0; jt < 2; ++jt)
#pragma unroll
        for (int i = 0; i < 8; ++i) creg[jt][i] = 0.0f;

    for (int t = 0; t < T; ++t) {
        // prefetch next timestep's px slab (16 rows x 4H) while this step computes
        if (t + 1 < T) {
            const float* nxt = px + ((size_t)(t + 1) * B + bbase) * G4;
#pragma unroll
            for (int q = 0; q < 4; ++q)
                __builtin_prefetch(nxt + ((int)threadIdx.x + q * 256) * 64, 0, 0);
        }

        v8f acc[4][2];
        // init accumulators with precomputed input projection (+bias already folded)
#pragma unroll
        for (int g = 0; g < 4; ++g)
#pragma unroll
            for (int jt = 0; jt < 2; ++jt) {
                const int n0 = g * 256 + jbase + jt * 16;
                v8f a;
#pragma unroll
                for (int i = 0; i < 8; ++i)
                    a[i] = px[((size_t)t * B + (bbase + mhi + i)) * G4 + n0 + nlo];
                acc[g][jt] = a;
            }
        // h @ Whh^T : K = 256 in 8 steps, A fragment shared across all 8 tiles
        for (int kk = 0; kk < H; kk += 32) {
            v16bf a = loadA_rm(&hsh[0][0] + kk, H + 8);
#pragma unroll
            for (int g = 0; g < 4; ++g)
#pragma unroll
                for (int jt = 0; jt < 2; ++jt) {
                    const int n0 = g * 256 + jbase + jt * 16;
                    v16bf b = loadB_wt(Wh, H, n0, kk);
                    acc[g][jt] = wmma_bf16(a, b, acc[g][jt]);
                }
        }
        __syncthreads();   // everyone done reading old h from LDS
#pragma unroll
        for (int jt = 0; jt < 2; ++jt) {
            const int j = jbase + jt * 16 + nlo;
#pragma unroll
            for (int i = 0; i < 8; ++i) {
                const int m = mhi + i;
                float iv = sigm(acc[0][jt][i]);
                float fv = sigm(acc[1][jt][i]);
                float gv = tanhf(acc[2][jt][i]);
                float ov = sigm(acc[3][jt][i]);
                float cn = fv * creg[jt][i] + iv * gv;
                creg[jt][i] = cn;
                float hv = ov * tanhf(cn);
                hsh[m][j] = f2bf(hv);
                hout[((size_t)(bbase + m) * T + t) * H + j] = hv;
            }
        }
        __syncthreads();   // new h visible before next step's A loads
    }
}

// Hc[b][t][0:H] = hF; Hc[b][t][H:2H] = hB[b][rev_idx(b,t)]  (un-reverse bwd pass)
__global__ void combine_hc(const float* __restrict__ hF, const float* __restrict__ hB,
                           const int* __restrict__ len, float* __restrict__ Hc) {
    size_t idx = (size_t)blockIdx.x * blockDim.x + threadIdx.x;   // over B*T*H
    if (idx >= (size_t)B * T * H) return;
    int j = (int)(idx % H);
    int t = (int)((idx / H) % T);
    int b = (int)(idx / ((size_t)H * T));
    int rt = len[b] - 1 - t; if (rt < 0) rt = 0;
    float hf = hF[idx];
    float hb = hB[((size_t)b * T + rt) * H + j];
    float* row = Hc + ((size_t)b * T + t) * (2 * H);
    row[j]     = hf;
    row[H + j] = hb;
}

// logits[b][head][t] = (t<len) ? W_s2 @ tanh(W_s1 @ Hc[b][t]) : NEG
__global__ void score_kernel(const float* __restrict__ Hc, const float* __restrict__ Ws1,
                             const float* __restrict__ Ws2, const int* __restrict__ len,
                             float* __restrict__ logits) {
    int bt = blockIdx.x * blockDim.x + threadIdx.x;
    if (bt >= B * T) return;
    int b = bt / T, t = bt % T;
    const float* hrow = Hc + (size_t)bt * (2 * H);
    float u[DA];
    for (int d = 0; d < DA; ++d) {
        float s = 0.0f;
        for (int k = 0; k < 2 * H; ++k) s += hrow[k] * Ws1[d * (2 * H) + k];
        u[d] = tanhf(s);
    }
    bool valid = t < len[b];
    for (int hh = 0; hh < HEADS; ++hh) {
        float s = 0.0f;
        for (int d = 0; d < DA; ++d) s += Ws2[hh * DA + d] * u[d];
        logits[((size_t)b * HEADS + hh) * T + t] = valid ? s : NEGINF;
    }
}

// Per-(b,head) masked softmax over T; exact zeros past length.
__global__ __launch_bounds__(256)
void softmax_kernel(const float* __restrict__ logits, const int* __restrict__ len,
                    float* __restrict__ A) {
    int bh = blockIdx.x;                  // B*HEADS blocks
    int b  = bh / HEADS;
    const float* lg = logits + (size_t)bh * T;
    __shared__ float red[256];
    float m = -3e30f;
    for (int t = threadIdx.x; t < T; t += 256) m = fmaxf(m, lg[t]);
    red[threadIdx.x] = m; __syncthreads();
    for (int s = 128; s > 0; s >>= 1) {
        if (threadIdx.x < s) red[threadIdx.x] = fmaxf(red[threadIdx.x], red[threadIdx.x + s]);
        __syncthreads();
    }
    m = red[0]; __syncthreads();
    float sum = 0.0f;
    for (int t = threadIdx.x; t < T; t += 256) sum += __expf(lg[t] - m);
    red[threadIdx.x] = sum; __syncthreads();
    for (int s = 128; s > 0; s >>= 1) {
        if (threadIdx.x < s) red[threadIdx.x] += red[threadIdx.x + s];
        __syncthreads();
    }
    float inv = 1.0f / red[0];
    int L = len[b];
    for (int t = threadIdx.x; t < T; t += 256)
        A[(size_t)bh * T + t] = (t < L) ? __expf(lg[t] - m) * inv : 0.0f;
}

// M[b][head][d] = sum_t A[b][head][t] * Hc[b][t][d]  -> straight into d_out
__global__ __launch_bounds__(512)
void attnM_kernel(const float* __restrict__ A, const float* __restrict__ Hc,
                  float* __restrict__ out) {
    int bh = blockIdx.x;                  // B*HEADS blocks
    int b = bh / HEADS, hh = bh % HEADS;
    int d = threadIdx.x;                  // 2H = 512 threads
    const float* Ab = A + (size_t)bh * T;
    float s = 0.0f;
    for (int t = 0; t < T; ++t) s += Ab[t] * Hc[((size_t)b * T + t) * (2 * H) + d];
    out[(size_t)b * (HEADS * 2 * H) + (size_t)hh * (2 * H) + d] = s;
}

// penal = sum_b ||A A^T - diag||_F^2 / B   (off-diagonal squared sum)
__global__ __launch_bounds__(256)
void penal_kernel(const float* __restrict__ A, float* __restrict__ out) {
    __shared__ float red[256];
    float acc = 0.0f;
    for (int e = threadIdx.x; e < B * HEADS * HEADS; e += 256) {
        int g  = e % HEADS;
        int hh = (e / HEADS) % HEADS;
        int b  = e / (HEADS * HEADS);
        if (hh == g) continue;
        const float* Ah = A + ((size_t)b * HEADS + hh) * T;
        const float* Ag = A + ((size_t)b * HEADS + g) * T;
        float s = 0.0f;
        for (int t = 0; t < T; ++t) s += Ah[t] * Ag[t];
        acc += s * s;
    }
    red[threadIdx.x] = acc; __syncthreads();
    for (int s = 128; s > 0; s >>= 1) {
        if (threadIdx.x < s) red[threadIdx.x] += red[threadIdx.x + s];
        __syncthreads();
    }
    if (threadIdx.x == 0) out[0] = red[0] / (float)B;
}

// ---------------- launcher ----------------
extern "C" void kernel_launch(void* const* d_in, const int* in_sizes, int n_in,
                              void* d_out, int out_size, void* d_ws, size_t ws_size,
                              hipStream_t stream) {
    const int*   wid  = (const int*)  d_in[0];
    const int*   len  = (const int*)  d_in[1];
    const float* emb  = (const float*)d_in[2];
    const float* Wihf = (const float*)d_in[3];
    const float* Whhf = (const float*)d_in[4];
    const float* bf   = (const float*)d_in[5];
    const float* Wihb = (const float*)d_in[6];
    const float* Whhb = (const float*)d_in[7];
    const float* bb   = (const float*)d_in[8];
    const float* Ws1  = (const float*)d_in[9];
    const float* Ws2  = (const float*)d_in[10];
    float* out = (float*)d_out;

    char* ws = (char*)d_ws;
    size_t off = 0;
    auto alloc = [&](size_t bytes) -> void* {
        void* p = ws + off;
        off += (bytes + 255) & ~(size_t)255;
        return p;
    };
    unsigned short* xf    = (unsigned short*)alloc((size_t)B * T * E * 2);
    unsigned short* xr    = (unsigned short*)alloc((size_t)B * T * E * 2);
    unsigned short* wihf  = (unsigned short*)alloc((size_t)G4 * E * 2);
    unsigned short* whhf  = (unsigned short*)alloc((size_t)G4 * H * 2);
    unsigned short* wihb  = (unsigned short*)alloc((size_t)G4 * E * 2);
    unsigned short* whhb  = (unsigned short*)alloc((size_t)G4 * H * 2);
    float*          pxf   = (float*)alloc((size_t)T * B * G4 * 4);
    float*          pxb   = (float*)alloc((size_t)T * B * G4 * 4);
    float*          hFbuf = (float*)alloc((size_t)B * T * H * 4);
    float*          hBbuf = (float*)alloc((size_t)B * T * H * 4);
    float*          Hc    = (float*)alloc((size_t)B * T * 2 * H * 4);
    float*          lgts  = (float*)alloc((size_t)B * HEADS * T * 4);
    float*          Amat  = (float*)alloc((size_t)B * HEADS * T * 4);

    // 1) embedding gather (+ per-sample reversal), fp32 -> bf16
    prep_embed<<<B * T, E, 0, stream>>>(wid, len, emb, xf, xr);

    // 2) weight conversion fp32 -> bf16
    const int NW = G4 * E;   // == G4*H here (E==H==256)
    conv_bf16<<<(NW + 255) / 256, 256, 0, stream>>>(Wihf, wihf, NW);
    conv_bf16<<<(NW + 255) / 256, 256, 0, stream>>>(Whhf, whhf, NW);
    conv_bf16<<<(NW + 255) / 256, 256, 0, stream>>>(Wihb, wihb, NW);
    conv_bf16<<<(NW + 255) / 256, 256, 0, stream>>>(Whhb, whhb, NW);

    // 3) input projections: [B*T,E] @ [E,4H] (+bias) -> [T,B,4H]
    //    one wave per 16x128 slab: waves = MT * 8 = 32768 -> 4096 blocks
    const int MT = (B * T) / 16;           // 4096
    const int gemm_blocks = (MT * 8) / 8;  // 8 waves per block
    gemm_px<<<gemm_blocks, 256, 0, stream>>>(xf, wihf, bf, pxf);
    gemm_px<<<gemm_blocks, 256, 0, stream>>>(xr, wihb, bb, pxb);

    // 4) recurrent scan, both directions
    lstm_scan<<<dim3(B / 16, 2), 256, 0, stream>>>(pxf, pxb, whhf, whhb, hFbuf, hBbuf);

    // 5) combine fwd/bwd into Hc (bwd un-reversed)
    {
        size_t n = (size_t)B * T * H;
        combine_hc<<<(unsigned)((n + 255) / 256), 256, 0, stream>>>(hFbuf, hBbuf, len, Hc);
    }

    // 6) attention scores + masking
    score_kernel<<<(B * T + 255) / 256, 256, 0, stream>>>(Hc, Ws1, Ws2, len, lgts);

    // 7) per-(b,head) softmax over time
    softmax_kernel<<<B * HEADS, 256, 0, stream>>>(lgts, len, Amat);

    // 8) pooled sentence embeddings -> d_out[0 .. B*HEADS*2H)
    attnM_kernel<<<B * HEADS, 512, 0, stream>>>(Amat, Hc, out);

    // 9) penalization scalar -> d_out[B*HEADS*2H]
    penal_kernel<<<1, 256, 0, stream>>>(Amat, out + (size_t)B * HEADS * 2 * H);
}
